// DeepseekV32Indexer_3341484556389
// MI455X (gfx1250) — matrix operand order repaired and checked by reading, not compiled, not run on hardware
//
#include <hip/hip_runtime.h>
#include <hip/hip_bf16.h>
#include <stdint.h>

// ---------------------------------------------------------------------------
// DeepSeek-V3.2 indexer for MI455X (gfx1250, wave32, WMMA).
// H=64 D=128 DR=64 S=T=2048 HID=7168 QLR=1536 TOPK=2048(==T -> full argsort)
// ---------------------------------------------------------------------------

#define S_LEN 2048
#define T_LEN 2048
#define HEADS 64
#define DHEAD 128
#define HID   7168
#define QLR   1536
#define QCOLS (HEADS * DHEAD) // 8192

typedef __attribute__((ext_vector_type(16))) __bf16          v16bf;
typedef __attribute__((ext_vector_type(8)))  float           v8f;
typedef __attribute__((ext_vector_type(8)))  unsigned short  ushort8;

union BfFrag {
    v16bf   bf;
    ushort8 u[2];
};

static __device__ __forceinline__ unsigned short f32_to_bf16(float f) {
    unsigned int u = __float_as_uint(f);
    u += 0x7FFFu + ((u >> 16) & 1u);     // round-to-nearest-even
    return (unsigned short)(u >> 16);
}
static __device__ __forceinline__ float bf16_to_f32(unsigned short h) {
    return __uint_as_float(((unsigned int)h) << 16);
}

static __device__ __forceinline__ v8f wmma_bf16(const BfFrag& a, const BfFrag& b, v8f c) {
    // D = A(16x32 bf16) * B(32x16 bf16) + C(16x16 f32)
    return __builtin_amdgcn_wmma_f32_16x16x32_bf16(
        /*neg_a=*/false, a.bf, /*neg_b=*/false, b.bf,
        /*c_mod=*/(short)0, c, /*reuse_a=*/false, /*reuse_b=*/false);
}

// ---------------------------------------------------------------------------
// 0) f32 -> bf16 conversion, vectorized (n must be a multiple of 8; it is).
// ---------------------------------------------------------------------------
__global__ void cvt_bf16_kernel(const float* __restrict__ src,
                                unsigned short* __restrict__ dst, int n8) {
    int i = blockIdx.x * blockDim.x + threadIdx.x;
    const int stride = gridDim.x * blockDim.x;
    for (; i < n8; i += stride) {
        const float4 a = ((const float4*)src)[i * 2];
        const float4 b = ((const float4*)src)[i * 2 + 1];
        ushort8 o;
        o[0] = f32_to_bf16(a.x); o[1] = f32_to_bf16(a.y);
        o[2] = f32_to_bf16(a.z); o[3] = f32_to_bf16(a.w);
        o[4] = f32_to_bf16(b.x); o[5] = f32_to_bf16(b.y);
        o[6] = f32_to_bf16(b.z); o[7] = f32_to_bf16(b.w);
        ((ushort8*)dst)[i] = o;
    }
}

// ---------------------------------------------------------------------------
// 1) K projection + LayerNorm + RoPE  (bf16 K out) and scaled head weights.
//    One block per token, 192 threads; hidden row staged in LDS (float4).
//    wsc = (hidden @ Ww) * H^-0.5 * D^-0.5   (both scales folded here)
// ---------------------------------------------------------------------------
__global__ void kproj_kernel(const float* __restrict__ HS,
                             const float* __restrict__ Wk,
                             const float* __restrict__ Ww,
                             const float* __restrict__ gamma,
                             const float* __restrict__ beta,
                             const float* __restrict__ cosd,
                             const float* __restrict__ sind,
                             unsigned short* __restrict__ K,
                             float* __restrict__ WSC) {
    __shared__ float hrow[HID];
    __shared__ float kbuf[DHEAD];
    __shared__ float knorm[DHEAD];

    const int tid = threadIdx.x;   // 0..191
    const int s   = blockIdx.x;

    const float4* src4 = (const float4*)(HS + (size_t)s * HID);
    for (int i = tid; i < HID / 4; i += 192) ((float4*)hrow)[i] = src4[i];
    __syncthreads();

    if (tid < DHEAD) {
        float acc = 0.f;
        #pragma unroll 4
        for (int i = 0; i < HID; ++i) acc = fmaf(hrow[i], Wk[(size_t)i * DHEAD + tid], acc);
        kbuf[tid] = acc;
    } else {
        const int h = tid - DHEAD;   // 0..63
        float acc = 0.f;
        #pragma unroll 4
        for (int i = 0; i < HID; ++i) acc = fmaf(hrow[i], Ww[(size_t)i * HEADS + h], acc);
        // H^-0.5 = 0.125 ; softmax_scale = 128^-0.5
        WSC[(size_t)s * HEADS + h] = acc * 0.125f * 0.088388347648318447f;
    }
    __syncthreads();

    if (tid < DHEAD) {
        float mu = 0.f;
        for (int i = 0; i < DHEAD; ++i) mu += kbuf[i];
        mu *= (1.f / 128.f);
        float var = 0.f;
        for (int i = 0; i < DHEAD; ++i) { float d = kbuf[i] - mu; var += d * d; }
        var *= (1.f / 128.f);
        knorm[tid] = (kbuf[tid] - mu) * rsqrtf(var + 1e-5f) * gamma[tid] + beta[tid];
    }
    __syncthreads();

    if (tid < 32) {
        const float xr = knorm[tid], xi = knorm[tid + 32];
        const float c  = cosd[s * 32 + tid], sn = sind[s * 32 + tid];
        K[(size_t)s * DHEAD + tid]      = f32_to_bf16(xr * c  - xi * sn);
        K[(size_t)s * DHEAD + tid + 32] = f32_to_bf16(xr * sn + xi * c);
    } else if (tid >= 64 && tid < DHEAD) {
        K[(size_t)s * DHEAD + tid] = f32_to_bf16(knorm[tid]);
    }
}

// ---------------------------------------------------------------------------
// 2) Q GEMM: Qraw[2048x8192] = qr[2048x1536] @ Wq[1536x8192]  (bf16 WMMA)
//    Block = 256 thr (8 waves). Block tile 64(m) x 128(n); wave = one
//    16(m) x 64(n) strip = 4 WMMA tiles sharing a single A fragment.
//    B chunk staged *transposed* in LDS: each thread loads two 16B rows of
//    B and writes the transpose as 8 packed b32 ds-stores (2 k per store).
// ---------------------------------------------------------------------------
__global__ void qgemm_kernel(const unsigned short* __restrict__ A,   // 2048x1536
                             const unsigned short* __restrict__ B,   // 1536x8192
                             unsigned short* __restrict__ C) {       // 2048x8192
    __shared__ unsigned short kbT[32 * 128];   // [n_local][k_local]  8 KB

    const int tid  = threadIdx.x;
    const int wave = tid >> 5;
    const int lane = tid & 31;
    const int lrow = lane & 15;
    const int hi   = lane >> 4;
    const int wsub = wave & 3;     // m sub-tile (0..3)
    const int whal = wave >> 2;    // n half (0..1), 64 cols each

    const int mBase  = blockIdx.y * 64;
    const int nBlock = blockIdx.x * 128;

    // staging assignment: 16 k-pairs x 16 n-chunks of 8
    const int kk2 = (tid >> 4) << 1;   // 0,2,..,30
    const int nn0 = (tid & 15) * 8;    // 0,8,..,120

    v8f acc[4];
    #pragma unroll
    for (int j = 0; j < 4; ++j) acc[j] = (v8f){0.f,0.f,0.f,0.f,0.f,0.f,0.f,0.f};

    const unsigned short* arow = A + (size_t)(mBase + wsub * 16 + lrow) * QLR;

    for (int k0 = 0; k0 < QLR; k0 += 32) {
        {
            const unsigned short* b0 = &B[(size_t)(k0 + kk2) * QCOLS + nBlock + nn0];
            const ushort8 r0 = *(const ushort8*)(b0);
            const ushort8 r1 = *(const ushort8*)(b0 + QCOLS);
            #pragma unroll
            for (int e = 0; e < 8; ++e) {
                const unsigned int packed =
                    (unsigned int)r0[e] | ((unsigned int)r1[e] << 16);
                *(unsigned int*)&kbT[(nn0 + e) * 32 + kk2] = packed;
            }
        }
        // prefetch next B chunk (global_prefetch_b8)
        if (k0 + 32 < QLR)
            __builtin_prefetch(&B[(size_t)(k0 + 32 + kk2) * QCOLS + nBlock + nn0], 0, 1);
        __syncthreads();

        BfFrag a;
        a.u[0] = *(const ushort8*)(arow + k0 + hi * 8);        // K: hi*8 .. +7
        a.u[1] = *(const ushort8*)(arow + k0 + 16 + hi * 8);   // K: 16+hi*8 .. +7
        #pragma unroll
        for (int j = 0; j < 4; ++j) {
            BfFrag b;
            const unsigned short* bp = &kbT[(whal * 64 + j * 16 + lrow) * 32 + hi * 8];
            b.u[0] = *(const ushort8*)(bp);
            b.u[1] = *(const ushort8*)(bp + 16);
            acc[j] = wmma_bf16(a, b, acc[j]);
        }
        __syncthreads();
    }

    #pragma unroll
    for (int j = 0; j < 4; ++j) {
        const int nCol = nBlock + whal * 64 + j * 16 + lrow;
        #pragma unroll
        for (int i = 0; i < 8; ++i) {
            const int m = mBase + wsub * 16 + i + hi * 8;
            C[(size_t)m * QCOLS + nCol] = f32_to_bf16(acc[j][i]);
        }
    }
}

// ---------------------------------------------------------------------------
// 3) In-place RoPE on Q's first 64 dims of each head (bf16).
// ---------------------------------------------------------------------------
__global__ void ropeq_kernel(unsigned short* __restrict__ Q,
                             const float* __restrict__ cosd,
                             const float* __restrict__ sind) {
    const int s = blockIdx.y;
    const int h = blockIdx.x;
    const int j = threadIdx.x;   // 0..31
    const size_t base = (size_t)s * QCOLS + h * DHEAD;
    const float xr = bf16_to_f32(Q[base + j]);
    const float xi = bf16_to_f32(Q[base + 32 + j]);
    const float c  = cosd[s * 32 + j], sn = sind[s * 32 + j];
    Q[base + j]      = f32_to_bf16(xr * c  - xi * sn);
    Q[base + 32 + j] = f32_to_bf16(xr * sn + xi * c);
}

// ---------------------------------------------------------------------------
// 4) index_scores[s,t] = sum_h relu( (q_sh . k_t) * wsc[s,h] )
//    Block tile: 64(s) x 128(t), 8 waves = 4 s-subtiles x 2 t-halves.
//    Wave owns a 16(s) x 64(t) strip (4 WMMA tiles, one shared A frag).
//    K tile (32 KB, flat ushort8 copy: same [t][d] layout as global) and
//    scaled weights (16 KB, float4 copy) in LDS. Per head: 16 chained
//    v_wmma_f32_16x16x32_bf16 per wave, then relu(c*w) accumulated in f32;
//    the 8 per-row w values are loaded once and reused across the 4 tiles.
// ---------------------------------------------------------------------------
__global__ void scores_kernel(const unsigned short* __restrict__ Q,   // 2048x8192
                              const unsigned short* __restrict__ K,   // 2048x128
                              const float* __restrict__ WSC,          // 2048x64
                              float* __restrict__ SC) {               // 2048x2048
    __shared__ unsigned short kb[128 * DHEAD];   // [t_local][d]   32 KB
    __shared__ float wb[64 * HEADS];             // [s_local][h]   16 KB

    const int tid  = threadIdx.x;
    const int wave = tid >> 5;
    const int lane = tid & 31;
    const int lrow = lane & 15;
    const int hi   = lane >> 4;
    const int wsub = wave & 3;     // s sub-tile (0..3)
    const int whal = wave >> 2;    // t half (0..1), 64 cols each

    const int s0 = blockIdx.y * 64;
    const int t0 = blockIdx.x * 128;

    // flat vector copies: kb rows are contiguous in global K
    {
        const ushort8* ksrc = (const ushort8*)(K + (size_t)t0 * DHEAD);
        for (int i = tid; i < (128 * DHEAD) / 8; i += 256) ((ushort8*)kb)[i] = ksrc[i];
        const float4* wsrc = (const float4*)(WSC + (size_t)s0 * HEADS);
        for (int i = tid; i < (64 * HEADS) / 4; i += 256) ((float4*)wb)[i] = wsrc[i];
    }
    __syncthreads();

    const int sr = s0 + wsub * 16 + lrow;        // A-matrix row for this lane
    float acc[4][8];
    #pragma unroll
    for (int j = 0; j < 4; ++j)
        #pragma unroll
        for (int i = 0; i < 8; ++i) acc[j][i] = 0.f;

    for (int h = 0; h < HEADS; ++h) {
        v8f c[4];
        #pragma unroll
        for (int j = 0; j < 4; ++j) c[j] = (v8f){0.f,0.f,0.f,0.f,0.f,0.f,0.f,0.f};

        const unsigned short* qrow = Q + (size_t)sr * QCOLS + h * DHEAD;
        #pragma unroll
        for (int k0 = 0; k0 < DHEAD; k0 += 32) {
            BfFrag a;
            a.u[0] = *(const ushort8*)(qrow + k0 + hi * 8);
            a.u[1] = *(const ushort8*)(qrow + k0 + 16 + hi * 8);
            #pragma unroll
            for (int j = 0; j < 4; ++j) {
                BfFrag b;
                const unsigned short* bp =
                    &kb[(whal * 64 + j * 16 + lrow) * DHEAD + k0 + hi * 8];
                b.u[0] = *(const ushort8*)(bp);
                b.u[1] = *(const ushort8*)(bp + 16);
                c[j] = wmma_bf16(a, b, c[j]);
            }
        }
        #pragma unroll
        for (int i = 0; i < 8; ++i) {
            const float w = wb[(wsub * 16 + i + hi * 8) * HEADS + h];
            #pragma unroll
            for (int j = 0; j < 4; ++j) {
                const float v = c[j][i] * w;
                acc[j][i] += (v > 0.f) ? v : 0.f;
            }
        }
    }

    #pragma unroll
    for (int j = 0; j < 4; ++j) {
        const int tc = t0 + whal * 64 + j * 16 + lrow;
        #pragma unroll
        for (int i = 0; i < 8; ++i)
            SC[(size_t)(s0 + wsub * 16 + i + hi * 8) * T_LEN + tc] = acc[j][i];
    }
}

// ---------------------------------------------------------------------------
// 5) Per-row full descending argsort (TOPK == T) via bitonic sort in LDS.
//    Comparator: value desc, tie -> lower index first (jax top_k semantics).
// ---------------------------------------------------------------------------
__global__ void sort_kernel(const float* __restrict__ SC, int* __restrict__ OUT) {
    __shared__ float v[2048];
    __shared__ int   id[2048];
    const int tid = threadIdx.x;   // 0..1023
    const int s   = blockIdx.x;
    const float* row = SC + (size_t)s * T_LEN;

    v[tid] = row[tid];               id[tid] = tid;
    v[tid + 1024] = row[tid + 1024]; id[tid + 1024] = tid + 1024;
    __syncthreads();

    for (int k = 2; k <= 2048; k <<= 1) {
        for (int j = k >> 1; j > 0; j >>= 1) {
            const int i = ((tid & ~(j - 1)) << 1) | (tid & (j - 1));
            const int p = i | j;
            const float va = v[i], vb = v[p];
            const int   ia = id[i], ib = id[p];
            const bool aFirst = (va > vb) || (va == vb && ia < ib);
            const bool desc   = ((i & k) == 0);
            if (desc ? !aFirst : aFirst) {
                v[i] = vb; v[p] = va; id[i] = ib; id[p] = ia;
            }
            __syncthreads();
        }
    }
    OUT[(size_t)s * T_LEN + tid]        = id[tid];
    OUT[(size_t)s * T_LEN + tid + 1024] = id[tid + 1024];
}

// ---------------------------------------------------------------------------
// Launch
// inputs: hidden(0) q_resid(1) cos(2) sin(3) Wq(4) Wk(5) gamma(6) beta(7) Ww(8)
// ---------------------------------------------------------------------------
extern "C" void kernel_launch(void* const* d_in, const int* in_sizes, int n_in,
                              void* d_out, int out_size, void* d_ws, size_t ws_size,
                              hipStream_t stream) {
    const float* hidden = (const float*)d_in[0];
    const float* qresid = (const float*)d_in[1];
    const float* cosd   = (const float*)d_in[2];
    const float* sind   = (const float*)d_in[3];
    const float* Wq     = (const float*)d_in[4];
    const float* Wk     = (const float*)d_in[5];
    const float* gamma  = (const float*)d_in[6];
    const float* beta   = (const float*)d_in[7];
    const float* Ww     = (const float*)d_in[8];

    char*  ws  = (char*)d_ws;
    size_t off = 0;
    auto alloc = [&](size_t bytes) -> void* {
        void* p = ws + off;
        off = (off + bytes + 255) & ~(size_t)255;
        return p;
    };

    unsigned short* qr16  = (unsigned short*)alloc((size_t)S_LEN * QLR * 2);     // 6 MB
    unsigned short* wq16  = (unsigned short*)alloc((size_t)QLR * QCOLS * 2);     // 24 MB
    unsigned short* q16   = (unsigned short*)alloc((size_t)S_LEN * QCOLS * 2);   // 32 MB
    unsigned short* k16   = (unsigned short*)alloc((size_t)T_LEN * DHEAD * 2);   // 0.5 MB
    float*          wsc   = (float*)alloc((size_t)S_LEN * HEADS * 4);            // 0.5 MB
    float*          sc    = (float*)alloc((size_t)S_LEN * T_LEN * 4);            // 16 MB

    cvt_bf16_kernel<<<2048, 256, 0, stream>>>(qresid, qr16, (S_LEN * QLR) / 8);
    cvt_bf16_kernel<<<4096, 256, 0, stream>>>(Wq, wq16, (QLR * QCOLS) / 8);

    kproj_kernel<<<S_LEN, 192, 0, stream>>>(hidden, Wk, Ww, gamma, beta,
                                            cosd, sind, k16, wsc);

    qgemm_kernel<<<dim3(QCOLS / 128, S_LEN / 64), 256, 0, stream>>>(qr16, wq16, q16);

    ropeq_kernel<<<dim3(HEADS, S_LEN), 32, 0, stream>>>(q16, cosd, sind);

    scores_kernel<<<dim3(T_LEN / 128, S_LEN / 64), 256, 0, stream>>>(q16, k16, wsc, sc);

    sort_kernel<<<S_LEN, 1024, 0, stream>>>(sc, (int*)d_out);
}
